// GCastHeterocoder_9191230013922
// MI455X (gfx1250) — matrix operand order repaired and checked by reading, not compiled, not run on hardware
//
#include <hip/hip_runtime.h>
#include <hip/hip_bf16.h>

#define DD 128

typedef __attribute__((ext_vector_type(16))) __bf16 v16bf;
typedef __attribute__((ext_vector_type(8)))  float  v8f;

union ABFrag {
    unsigned u[8];
    uint4    q[2];
    v16bf    v;
};

__device__ __forceinline__ unsigned pack_bf16x2(float a, float b) {
    unsigned ua = __float_as_uint(a), ub = __float_as_uint(b);
    ua += 0x7FFFu + ((ua >> 16) & 1u);   // round-to-nearest-even
    ub += 0x7FFFu + ((ub >> 16) & 1u);
    return (ua >> 16) | (ub & 0xFFFF0000u);
}

__device__ __forceinline__ float silu_f(float x) {
    return x / (1.0f + __expf(-x));
}

// One 16-row x 128-col tile GEMM: A staged in LDS as bf16 pairs (row stride SW
// 32-bit words), B = packed bf16 weights, row-major [128][KT] (KT/2 words/row).
// Wave wv (0..7) produces output columns 16*wv .. 16*wv+15.
// Layouts per CDNA5 ISA 7.12.2:
//   A 16x32 bf16: lanes 0-15 row m hold K 0-7 (V0-3) & 16-23 (V4-7); lanes 16-31 hold K 8-15 & 24-31.
//   B 32x16 bf16: lanes 0-15 col n hold K 0-7 (V0-3) & 16-23 (V4-7); lanes 16-31 hold K 8-15 & 24-31.
template <int KT>
__device__ __forceinline__ v8f tile_matmul(const unsigned* __restrict__ smem, int SW,
                                           const unsigned* __restrict__ wbf) {
    const int l    = threadIdx.x & 31;
    const int wv   = threadIdx.x >> 5;
    const int m    = l & 15;
    const int half = l >> 4;
    const int ncol = (wv << 4) + m;

    const unsigned* arow = smem + m * SW + half * 4;
    const unsigned* brow = wbf + ncol * (KT / 2) + half * 4;

    v8f acc = {};
#pragma unroll
    for (int kc = 0; kc < KT / 32; ++kc) {
        const int kbw = kc << 4;           // 16 words per 32-K chunk
        ABFrag a, b;
        a.q[0] = *(const uint4*)(arow + kbw);       // K: kb+half*8 .. +7
        a.q[1] = *(const uint4*)(arow + kbw + 8);   // K: kb+16+half*8 .. +7
        b.q[0] = *(const uint4*)(brow + kbw);       // K: kb+half*8 .. +7
        b.q[1] = *(const uint4*)(brow + kbw + 8);   // K: kb+16+half*8 .. +7
        acc = __builtin_amdgcn_wmma_f32_16x16x32_bf16(
            false, a.v, false, b.v, (short)0, acc, false, false);
    }
    return acc;
}

// Activation epilogue: z = acc + bias; y = silu(silu(z)); store y to LDS [16][128].
__device__ __forceinline__ void act_to_lds(const v8f& acc, const float* __restrict__ bias,
                                           float* __restrict__ ybuf) {
    const int l    = threadIdx.x & 31;
    const int wv   = threadIdx.x >> 5;
    const int half = l >> 4;
    const int c    = (wv << 4) + (l & 15);
    const float bz = bias[c];
#pragma unroll
    for (int r = 0; r < 8; ++r) {
        const int mm = r + (half << 3);
        float z = acc[r] + bz;
        ybuf[mm * DD + c] = silu_f(silu_f(z));
    }
}

// LayerNorm over 128 cols of row (threadIdx.x>>4); this thread owns 8 cols.
// Returns rs, mu and fills yv[8].
__device__ __forceinline__ void ln_reduce(const float* __restrict__ ybuf,
                                          float yv[8], float& mu, float& rs) {
    const int row = threadIdx.x >> 4;
    const int cg  = threadIdx.x & 15;
    float s = 0.f, s2 = 0.f;
#pragma unroll
    for (int j = 0; j < 8; ++j) {
        float v = ybuf[row * DD + cg * 8 + j];
        yv[j] = v; s += v; s2 += v * v;
    }
#pragma unroll
    for (int msk = 1; msk <= 8; msk <<= 1) {  // 16-lane groups stay in wave halves
        s  += __shfl_xor(s,  msk);
        s2 += __shfl_xor(s2, msk);
    }
    mu = s * (1.0f / 128.0f);
    float var = s2 * (1.0f / 128.0f) - mu * mu;
    rs = rsqrtf(var + 1e-5f);
}

// ---------- weight fp32 -> packed bf16 conversion ----------
__global__ __launch_bounds__(256) void convert_weights_kernel(
    const float* __restrict__ We, const float* __restrict__ Wn, const float* __restrict__ Ws,
    unsigned* __restrict__ we_bf, unsigned* __restrict__ wn_bf, unsigned* __restrict__ ws_bf) {
    int w = blockIdx.x * 256 + threadIdx.x;
    const int NWE = DD * 3 * DD / 2;   // 24576
    const int NWN = DD * 2 * DD / 2;   // 16384
    const int NWS = DD * DD / 2;       // 8192
    if (w < NWE) {
        we_bf[w] = pack_bf16x2(We[2 * w], We[2 * w + 1]);
    } else if (w < NWE + NWN) {
        int i = w - NWE;
        wn_bf[i] = pack_bf16x2(Wn[2 * i], Wn[2 * i + 1]);
    } else if (w < NWE + NWN + NWS) {
        int i = w - NWE - NWN;
        ws_bf[i] = pack_bf16x2(Ws[2 * i], Ws[2 * i + 1]);
    }
}

// ---------- edge encoder: gather+concat -> MLP -> LN -> residual + scatter-sum ----------
__global__ __launch_bounds__(256) void edge_kernel(
    const float* __restrict__ sender_x, const float* __restrict__ receiver_x,
    const int* __restrict__ eidx, const float* __restrict__ edge_attr,
    const float* __restrict__ be, const float* __restrict__ ge, const float* __restrict__ bbe,
    const unsigned* __restrict__ we_bf, float* __restrict__ aggr,
    float* __restrict__ edge_out, int E, int NR) {
    constexpr int KT = 3 * DD;      // 384
    constexpr int SW = KT / 2 + 4;  // 196 words, padded
    __shared__ unsigned sin[16 * SW];
    __shared__ float    ybuf[16 * DD];

    const int  t   = threadIdx.x;
    const int  r   = t >> 4;
    const int  l16 = t & 15;
    const long e   = (long)blockIdx.x * 16 + r;
    const long ec  = e < E ? e : (long)E - 1;
    const int  src = eidx[ec];
    const int  dst = eidx[(long)E + ec];

    const float* s_row = sender_x  + (long)src * DD;
    const float* r_row = receiver_x + (long)dst * DD;
    const float* e_row = edge_attr + ec * DD;
#pragma unroll
    for (int i = 0; i < 12; ++i) {
        int pair = l16 + (i << 4);     // 0..191
        int k = pair << 1;
        float a, b;
        if (k < DD)            { a = s_row[k];          b = s_row[k + 1]; }
        else if (k < 2 * DD)   { a = r_row[k - DD];     b = r_row[k - DD + 1]; }
        else                   { a = e_row[k - 2 * DD]; b = e_row[k - 2 * DD + 1]; }
        sin[r * SW + pair] = pack_bf16x2(a, b);
    }
    __syncthreads();

    v8f acc = tile_matmul<KT>(sin, SW, we_bf);
    act_to_lds(acc, be, ybuf);
    __syncthreads();

    float yv[8], mu, rs;
    ln_reduce(ybuf, yv, mu, rs);

    if (e < E) {
        const int cg = t & 15;
        const float* erow = edge_attr + e * DD;
        float* orow = edge_out + e * DD;
        float* arow = aggr + (long)dst * DD;
#pragma unroll
        for (int j = 0; j < 8; ++j) {
            int c = cg * 8 + j;
            float p = (yv[j] - mu) * rs * ge[c] + bbe[c];
            orow[c] = erow[c] + p;                // edge residual
            unsafeAtomicAdd(arow + c, p);         // scatter-sum to receiver
        }
    }
}

// ---------- node encoder: [receiver_x | aggr] -> MLP -> LN -> residual ----------
__global__ __launch_bounds__(256) void node_kernel(
    const float* __restrict__ receiver_x, const float* __restrict__ aggr,
    const float* __restrict__ bn, const float* __restrict__ gn, const float* __restrict__ bbn,
    const unsigned* __restrict__ wn_bf, float* __restrict__ recv_out, int NR) {
    constexpr int KT = 2 * DD;      // 256
    constexpr int SW = KT / 2 + 4;  // 132
    __shared__ unsigned sin[16 * SW];
    __shared__ float    ybuf[16 * DD];

    const int  t   = threadIdx.x;
    const int  r   = t >> 4;
    const int  l16 = t & 15;
    const long n   = (long)blockIdx.x * 16 + r;
    const long nc  = n < NR ? n : (long)NR - 1;

    const float* x_row = receiver_x + nc * DD;
    const float* a_row = aggr + nc * DD;
#pragma unroll
    for (int i = 0; i < 8; ++i) {
        int pair = l16 + (i << 4);     // 0..127
        int k = pair << 1;
        float a, b;
        if (k < DD) { a = x_row[k];      b = x_row[k + 1]; }
        else        { a = a_row[k - DD]; b = a_row[k - DD + 1]; }
        sin[r * SW + pair] = pack_bf16x2(a, b);
    }
    __syncthreads();

    v8f acc = tile_matmul<KT>(sin, SW, wn_bf);
    act_to_lds(acc, bn, ybuf);
    __syncthreads();

    float yv[8], mu, rs;
    ln_reduce(ybuf, yv, mu, rs);

    if (n < NR) {
        const int cg = t & 15;
        const float* xr = receiver_x + n * DD;
        float* orow = recv_out + n * DD;
#pragma unroll
        for (int j = 0; j < 8; ++j) {
            int c = cg * 8 + j;
            float p = (yv[j] - mu) * rs * gn[c] + bbn[c];
            orow[c] = xr[c] + p;
        }
    }
}

// ---------- sender MLP: square D->D, residual ----------
__global__ __launch_bounds__(256) void sender_kernel(
    const float* __restrict__ sender_x,
    const float* __restrict__ bs, const float* __restrict__ gs, const float* __restrict__ bbs,
    const unsigned* __restrict__ ws_bf, float* __restrict__ send_out, int NS) {
    constexpr int KT = DD;          // 128
    constexpr int SW = KT / 2 + 4;  // 68
    __shared__ unsigned sin[16 * SW];
    __shared__ float    ybuf[16 * DD];

    const int  t   = threadIdx.x;
    const int  r   = t >> 4;
    const int  l16 = t & 15;
    const long n   = (long)blockIdx.x * 16 + r;
    const long nc  = n < NS ? n : (long)NS - 1;

    const float* x_row = sender_x + nc * DD;
#pragma unroll
    for (int i = 0; i < 4; ++i) {
        int pair = l16 + (i << 4);     // 0..63
        int k = pair << 1;
        sin[r * SW + pair] = pack_bf16x2(x_row[k], x_row[k + 1]);
    }
    __syncthreads();

    v8f acc = tile_matmul<KT>(sin, SW, ws_bf);
    act_to_lds(acc, bs, ybuf);
    __syncthreads();

    float yv[8], mu, rs;
    ln_reduce(ybuf, yv, mu, rs);

    if (n < NS) {
        const int cg = t & 15;
        const float* xr = sender_x + n * DD;
        float* orow = send_out + n * DD;
#pragma unroll
        for (int j = 0; j < 8; ++j) {
            int c = cg * 8 + j;
            float p = (yv[j] - mu) * rs * gs[c] + bbs[c];
            orow[c] = xr[c] + p;
        }
    }
}

extern "C" void kernel_launch(void* const* d_in, const int* in_sizes, int n_in,
                              void* d_out, int out_size, void* d_ws, size_t ws_size,
                              hipStream_t stream) {
    const float* sender_x   = (const float*)d_in[0];
    const float* receiver_x = (const float*)d_in[1];
    const int*   eidx       = (const int*)d_in[2];
    const float* edge_attr  = (const float*)d_in[3];
    const float* We  = (const float*)d_in[4];
    const float* be  = (const float*)d_in[5];
    const float* ge  = (const float*)d_in[6];
    const float* bbe = (const float*)d_in[7];
    const float* Wn  = (const float*)d_in[8];
    const float* bn  = (const float*)d_in[9];
    const float* gn  = (const float*)d_in[10];
    const float* bbn = (const float*)d_in[11];
    const float* Ws  = (const float*)d_in[12];
    const float* bs  = (const float*)d_in[13];
    const float* gs  = (const float*)d_in[14];
    const float* bbs = (const float*)d_in[15];

    const int NS = in_sizes[0] / DD;
    const int NR = in_sizes[1] / DD;
    const int E  = in_sizes[2] / 2;

    float* out_s = (float*)d_out;
    float* out_r = out_s + (size_t)NS * DD;
    float* out_e = out_r + (size_t)NR * DD;

    // workspace layout: [aggr NR*D f32][We_bf][Wn_bf][Ws_bf]
    char* wsb = (char*)d_ws;
    float* aggr = (float*)wsb;
    size_t aggr_bytes = (size_t)NR * DD * sizeof(float);
    unsigned* we_bf = (unsigned*)(wsb + aggr_bytes);
    unsigned* wn_bf = we_bf + DD * 3 * DD / 2;
    unsigned* ws_bf = wn_bf + DD * 2 * DD / 2;

    // 1) zero scatter accumulator (graph-capture legal)
    hipMemsetAsync(aggr, 0, aggr_bytes, stream);

    // 2) weights -> bf16
    {
        int total = DD * 3 * DD / 2 + DD * 2 * DD / 2 + DD * DD / 2;  // 49152 words
        convert_weights_kernel<<<(total + 255) / 256, 256, 0, stream>>>(
            We, Wn, Ws, we_bf, wn_bf, ws_bf);
    }

    // 3) edge MLP + scatter-sum (must precede node kernel; stream-ordered)
    edge_kernel<<<(E + 15) / 16, 256, 0, stream>>>(
        sender_x, receiver_x, eidx, edge_attr, be, ge, bbe, we_bf, aggr, out_e, E, NR);

    // 4) node MLP (+residual)
    node_kernel<<<(NR + 15) / 16, 256, 0, stream>>>(
        receiver_x, aggr, bn, gn, bbn, wn_bf, out_r, NR);

    // 5) sender MLP (+residual) — independent
    sender_kernel<<<(NS + 15) / 16, 256, 0, stream>>>(
        sender_x, bs, gs, bbs, ws_bf, out_s, NS);
}
